// DA_RNN_73504070304224
// MI455X (gfx1250) — matrix operand order repaired and checked by reading, not compiled, not run on hardware
//
#include <hip/hip_runtime.h>
#include <hip/hip_bf16.h>
#include <math.h>

// DA-RNN fused persistent kernel for MI455X (gfx1250, wave32).
// 16 workgroups x 16 batch rows; whole S=512 recurrence on-chip.
// bf16 WMMA (v_wmma_f32_16x16x32_bf16) with fp32 accumulate.
// Weights are bf16-prepacked into WMMA B-fragment layout and streamed
// from L2 every step (688 KB total, L2-resident); an opaque-zero offset
// defeats LICM so the fragment loads are NOT hoisted+spilled to scratch.
// All transcendentals use hardware v_exp_f32 / v_rcp_f32 (no libm branches).

typedef __bf16 v16bf __attribute__((ext_vector_type(16)));
typedef float  v8f   __attribute__((ext_vector_type(8)));

#define S_   512
#define I_   128
#define H_   256

// Opaque 0: compiler cannot fold it, so weight addresses look t-varying and
// the per-step global_load_b128s stay inside the recurrence loop (L2 hits)
// instead of being hoisted and spilled to scratch.
__device__ __forceinline__ int opaque_zero() {
  int z;
  asm volatile("s_mov_b32 %0, 0" : "=s"(z));
  return z;
}

// Straight-line transcendentals on the v_exp_f32 / v_rcp_f32 TRANS pipes.
__device__ __forceinline__ float fast_exp(float x) {
  return __builtin_amdgcn_exp2f(x * 1.44269504088896f);
}
__device__ __forceinline__ float fast_rcp(float x) {
  return __builtin_amdgcn_rcpf(x);
}
__device__ __forceinline__ float fast_sigmoid(float x) {
  return fast_rcp(1.f + fast_exp(-x));
}
__device__ __forceinline__ float fast_tanh(float x) {
  // tanh(x) = 1 - 2/(exp(2x)+1); exp->inf gives +1, exp->0 gives -1.
  return 1.f - 2.f * fast_rcp(fast_exp(2.f * x) + 1.f);
}

__device__ __forceinline__ v16bf frag_pair(const __bf16* p0, const __bf16* p1) {
  union { v16bf v; uint4 q[2]; } u;
  u.q[0] = *(const uint4*)p0;
  u.q[1] = *(const uint4*)p1;
  return u.v;
}

// A-fragment (16x32 bf16, ISA 7.12.2): lane m=lane&15, g=lane>>4.
// halves 0..7 -> k = g*8 + 0..7 ; halves 8..15 -> k = g*8 + 16..23.
// rowp points at row m, column kt*32 of a row-major bf16 LDS tile.
__device__ __forceinline__ v16bf load_a(const __bf16* rowp, int g) {
  const __bf16* p = rowp + g * 8;
  return frag_pair(p, p + 16);
}

// B-fragment (32x16 bf16): pre-packed lane-major (32 lanes x 16 halves).
__device__ __forceinline__ v16bf load_b(const __bf16* fragbase, int lane) {
  const __bf16* p = fragbase + lane * 16;
  return frag_pair(p, p + 8);
}

__device__ __forceinline__ v8f wmma_bf16(v16bf a, v16bf b, v8f c) {
  return __builtin_amdgcn_wmma_f32_16x16x32_bf16(false, a, false, b, (short)0, c,
                                                 false, false);
}

__device__ __forceinline__ float redsum16(float v) {
  #pragma unroll
  for (int off = 8; off; off >>= 1) v += __shfl_xor(v, off, 16);
  return v;
}
__device__ __forceinline__ float redmax16(float v) {
  #pragma unroll
  for (int off = 8; off; off >>= 1) v = fmaxf(v, __shfl_xor(v, off, 16));
  return v;
}

// Pre-pack weight W [N][K] (row-major fp32) acting as B-matrix B(k,n)=W[n,k]
// into bf16 WMMA B-fragments: frag = kt*(N/16)+nt, lane-major, 16 halves/lane.
// B layout (ISA 7.12.2): lane half h, halves 0..7 -> k=8h+0..7, halves 8..15 -> k=16+8h+0..7.
__global__ void pack_b(const float* __restrict__ W, __bf16* __restrict__ out,
                       int N, int K) {
  int tid = blockIdx.x * 256 + threadIdx.x;
  if (tid >= N * K) return;
  int frag   = tid >> 9;        // /512 halves per fragment
  int within = tid & 511;
  int lane   = within >> 4;
  int j      = within & 15;
  int Nt = N >> 4;
  int kt = frag / Nt, nt = frag % Nt;
  int k = kt * 32 + 8 * (lane >> 4) + ((j < 8) ? j : 8 + j);
  int n = nt * 16 + (lane & 15);
  out[tid] = (__bf16)W[(size_t)n * K + k];
}

// gi = xw @ W_ih^T (K=128, 4 k-tiles), gh = h @ W_hh^T (K=256, 8 k-tiles)
// for one 16x16 output tile `tile` (0..47 across 3H=768 columns).
__device__ __forceinline__ void gate_gemm(const __bf16* s_xw, const __bf16* s_h16,
                                          const __bf16* wih, const __bf16* whh,
                                          int tile, int am, int g, int lane,
                                          v8f& gi, v8f& gh) {
  #pragma unroll
  for (int kt = 0; kt < 4; ++kt) {
    v16bf a = load_a(s_xw + am * I_ + kt * 32, g);
    v16bf b = load_b(wih + (size_t)(kt * 48 + tile) * 512, lane);
    gi = wmma_bf16(a, b, gi);
  }
  #pragma unroll
  for (int kt = 0; kt < 8; ++kt) {
    v16bf a = load_a(s_h16 + am * H_ + kt * 32, g);
    v16bf b = load_b(whh + (size_t)(kt * 48 + tile) * 512, lane);
    gh = wmma_bf16(a, b, gh);
  }
}

__global__ __launch_bounds__(256, 1) void darnn_persistent(
    const float* __restrict__ x,
    const float* __restrict__ b_a,
    const float* __restrict__ b_ih,
    const float* __restrict__ b_hh,
    const float* __restrict__ W_t,
    const float* __restrict__ b_t,
    const float* __restrict__ W_f,
    const float* __restrict__ b_f,
    const __bf16* __restrict__ wa_g,
    const __bf16* __restrict__ wih_g,
    const __bf16* __restrict__ whh_g,
    float* __restrict__ out) {
  extern __shared__ char smem[];
  float*  s_h    = (float*)smem;                 // [16][256] fp32 hidden
  float*  s_ctx  = s_h   + 16 * 256;             // [16][256] running context
  float*  s_r    = s_ctx + 16 * 256;             // [16][256] reset gate
  float*  s_z    = s_r   + 16 * 256;             // [16][256] update gate
  float*  s_sc   = s_z   + 16 * 256;             // [16][128] attn scores
  __bf16* s_h16  = (__bf16*)(s_sc + 16 * 128);   // [16][256] bf16 hidden
  __bf16* s_x    = s_h16 + 16 * 256;             // [16][128] bf16 x_t
  __bf16* s_xw   = s_x   + 16 * 128;             // [16][128] bf16 alpha*x_t
  float*  s_ts   = (float*)(s_xw + 16 * 128);    // [16] temporal score
  float*  s_mrun = s_ts   + 16;                  // [16] running max
  float*  s_srun = s_mrun + 16;                  // [16] running sum
  float*  s_corr = s_srun + 16;                  // [16] correction
  float*  s_p    = s_corr + 16;                  // [16] new weight

  const int tid  = threadIdx.x;
  const int lane = tid & 31;
  const int wid  = tid >> 5;          // 8 waves
  const int g    = (lane >> 4) & 1;
  const int am   = lane & 15;
  const int wg   = blockIdx.x;        // owns batch rows wg*16 .. wg*16+15

  for (int i = tid; i < 16 * 256; i += 256) {
    s_h[i] = 0.f; s_ctx[i] = 0.f; s_h16[i] = (__bf16)0.f;
  }
  if (tid < 16) { s_mrun[tid] = -3.4e38f; s_srun[tid] = 0.f; }
  __syncthreads();

  for (int t = 0; t < S_; ++t) {
    // Opaque offset: forces all weight-fragment loads to stay in-loop.
    const int oz = opaque_zero();
    const __bf16* wa  = wa_g  + oz;
    const __bf16* wih = wih_g + oz;
    const __bf16* whh = whh_g + oz;

    // ---- stage x_t into LDS (bf16), prefetch x_{t+1} ----
    #pragma unroll
    for (int k = 0; k < 8; ++k) {
      int idx = tid * 8 + k;
      int m = idx >> 7, i = idx & 127;
      float xv = x[((size_t)(wg * 16 + m)) * S_ * I_ + (size_t)t * I_ + i];
      s_x[m * I_ + i] = (__bf16)xv;
    }
    if (t + 1 < S_) {
      int pm = tid >> 4, pc = (tid & 15) * 8;
      __builtin_prefetch(
          &x[((size_t)(wg * 16 + pm)) * S_ * I_ + (size_t)(t + 1) * I_ + pc], 0, 3);
    }
    __syncthreads();

    // ---- phase 1: scores = tanh([x_t,h] @ W_a^T + b_a); wave w -> N-tile w ----
    {
      v8f acc = {};
      #pragma unroll
      for (int kt = 0; kt < 12; ++kt) {
        const __bf16* src = (kt < 4) ? (s_x + am * I_ + kt * 32)
                                     : (s_h16 + am * H_ + (kt - 4) * 32);
        v16bf a = load_a(src, g);
        v16bf b = load_b(wa + (size_t)(kt * 8 + wid) * 512, lane);
        acc = wmma_bf16(a, b, acc);
      }
      int n = wid * 16 + am;
      float bias = b_a[n];
      #pragma unroll
      for (int v = 0; v < 8; ++v)
        s_sc[(v + 8 * g) * I_ + n] = fast_tanh(acc[v] + bias);
    }
    __syncthreads();

    // ---- phase 2: row softmax over I, xw = alpha * x_t ----
    {
      int row = wid * 2 + g;
      int c0 = am * 8;
      float mx = -3.4e38f;
      #pragma unroll
      for (int j = 0; j < 8; ++j) mx = fmaxf(mx, s_sc[row * I_ + c0 + j]);
      mx = redmax16(mx);
      float se = 0.f;
      #pragma unroll
      for (int j = 0; j < 8; ++j) se += fast_exp(s_sc[row * I_ + c0 + j] - mx);
      se = redsum16(se);
      float inv = fast_rcp(se);
      #pragma unroll
      for (int j = 0; j < 8; ++j) {
        int c = c0 + j;
        float alpha = fast_exp(s_sc[row * I_ + c] - mx) * inv;
        s_xw[row * I_ + c] = (__bf16)(alpha * (float)s_x[row * I_ + c]);
      }
    }
    __syncthreads();

    // ---- phase 3: gate GEMMs + GRU update ----
    // wave w owns tiles {w+8q}: q=0,1 -> r ; q=2,3 -> z ; q=4,5 -> n
    v8f ni0 = {}, nh0 = {}, ni1 = {}, nh1 = {};
    #pragma unroll
    for (int q = 0; q < 4; ++q) {
      int tile = wid + 8 * q;
      v8f gi = {}, gh = {};
      gate_gemm(s_xw, s_h16, wih, whh, tile, am, g, lane, gi, gh);
      int nglob = tile * 16 + am;
      float bi = b_ih[nglob], bh = b_hh[nglob];
      int c = nglob & 255;
      float* dst = (q < 2) ? s_r : s_z;
      #pragma unroll
      for (int v = 0; v < 8; ++v) {
        float val = gi[v] + bi + gh[v] + bh;
        dst[(v + 8 * g) * H_ + c] = fast_sigmoid(val);
      }
    }
    // n-gate GEMMs: all reads of s_h16 finish BEFORE the barrier; accumulators
    // stay in registers across it so the in-place h update is race-free.
    gate_gemm(s_xw, s_h16, wih, whh, wid + 32, am, g, lane, ni0, nh0);
    gate_gemm(s_xw, s_h16, wih, whh, wid + 40, am, g, lane, ni1, nh1);
    __syncthreads();
    #pragma unroll
    for (int q = 0; q < 2; ++q) {
      int tile = wid + 32 + 8 * q;
      v8f gi = q ? ni1 : ni0;
      v8f gh = q ? nh1 : nh0;
      int nglob = tile * 16 + am;
      float bi = b_ih[nglob], bh = b_hh[nglob];
      int c = nglob & 255;
      #pragma unroll
      for (int v = 0; v < 8; ++v) {
        int mo = v + 8 * g;
        float r  = s_r[mo * H_ + c];
        float z  = s_z[mo * H_ + c];
        float nv = fast_tanh(gi[v] + bi + r * (gh[v] + bh));
        float hn = (1.f - z) * nv + z * s_h[mo * H_ + c];
        s_h[mo * H_ + c]   = hn;
        s_h16[mo * H_ + c] = (__bf16)hn;
      }
    }
    __syncthreads();

    // ---- phase 4: online temporal attention (flash-style running softmax) ----
    {
      int row = wid * 2 + g;
      int c0 = am * 16;
      float part = 0.f;
      #pragma unroll
      for (int j = 0; j < 16; ++j) part += s_h[row * H_ + c0 + j] * W_t[c0 + j];
      part = redsum16(part);
      if (am == 0) s_ts[row] = fast_tanh(part + b_t[0]);
    }
    __syncthreads();
    if (tid < 16) {
      float s    = s_ts[tid];
      float mo   = s_mrun[tid];
      float mn   = fmaxf(mo, s);
      float corr = fast_exp(mo - mn);
      float p    = fast_exp(s - mn);
      s_srun[tid] = s_srun[tid] * corr + p;
      s_mrun[tid] = mn;
      s_corr[tid] = corr;
      s_p[tid]    = p;
    }
    __syncthreads();
    {
      int rm = tid >> 4, cc = (tid & 15) * 16;
      float corr = s_corr[rm], p = s_p[rm];
      #pragma unroll
      for (int j = 0; j < 16; ++j) {
        int idx = rm * H_ + cc + j;
        s_ctx[idx] = s_ctx[idx] * corr + p * s_h[idx];
      }
    }
    __syncthreads();
  }

  // ---- epilogue: sigmoid((ctx/srun) @ W_f^T + b_f) ----
  {
    int row = wid * 2 + g;
    float inv = fast_rcp(s_srun[row]);
    int c0 = am * 16;
    float part = 0.f;
    #pragma unroll
    for (int j = 0; j < 16; ++j) part += s_ctx[row * H_ + c0 + j] * W_f[c0 + j];
    part = redsum16(part);
    if (am == 0)
      out[wg * 16 + row] = fast_sigmoid(part * inv + b_f[0]);
  }
}

extern "C" void kernel_launch(void* const* d_in, const int* in_sizes, int n_in,
                              void* d_out, int out_size, void* d_ws, size_t ws_size,
                              hipStream_t stream) {
  const float* x    = (const float*)d_in[0];
  const float* W_a  = (const float*)d_in[1];
  const float* b_a  = (const float*)d_in[2];
  const float* W_ih = (const float*)d_in[3];
  const float* b_ih = (const float*)d_in[4];
  const float* W_hh = (const float*)d_in[5];
  const float* b_hh = (const float*)d_in[6];
  const float* W_t  = (const float*)d_in[7];
  const float* b_t  = (const float*)d_in[8];
  const float* W_f  = (const float*)d_in[9];
  const float* b_f  = (const float*)d_in[10];
  float* out = (float*)d_out;

  __bf16* wa  = (__bf16*)d_ws;           // 128*384 halves
  __bf16* wih = wa  + 128 * 384;         // 768*128 halves
  __bf16* whh = wih + 768 * 128;         // 768*256 halves

  pack_b<<<(128 * 384 + 255) / 256, 256, 0, stream>>>(W_a,  wa,  128, 384);
  pack_b<<<(768 * 128 + 255) / 256, 256, 0, stream>>>(W_ih, wih, 768, 128);
  pack_b<<<(768 * 256 + 255) / 256, 256, 0, stream>>>(W_hh, whh, 768, 256);

  size_t shmem = (size_t)(16 * 256 * 4) * 4   // h, ctx, r, z (fp32)
               + (size_t)(16 * 128 * 4)       // scores (fp32)
               + (size_t)(16 * 256 + 2 * 16 * 128) * 2  // h16, x, xw (bf16)
               + 16 * 5 * 4 + 64;             // per-row scalars + pad
  (void)hipFuncSetAttribute((const void*)darnn_persistent,
                            hipFuncAttributeMaxDynamicSharedMemorySize,
                            (int)shmem);
  darnn_persistent<<<16, 256, shmem, stream>>>(x, b_a, b_ih, b_hh, W_t, b_t,
                                               W_f, b_f, wa, wih, whh, out);
}
